// PrediNet_73572789781129
// MI455X (gfx1250) — matrix operand order repaired and checked by reading, not compiled, not run on hardware
//
#include <hip/hip_runtime.h>
#include <hip/hip_bf16.h>

typedef __attribute__((ext_vector_type(16))) __bf16 v16bf;
typedef __attribute__((ext_vector_type(8)))  float  v8f;

#define BGRAPHS 64
#define DDIM    256
#define HEADS   16
#define KDIM    32
#define RDIM    16
#define MAXN    2048
#define CHUNK   128    // nodes staged per LDS tile
#define XTS     136    // transposed-stage row stride (128 + 8 pad: bank spread, 16B align)

__device__ __forceinline__ v8f wmma_bf(v16bf a, v16bf b, v8f c) {
  return __builtin_amdgcn_wmma_f32_16x16x32_bf16(false, a, false, b, (short)0, c, false, false);
}

// ---- Kernel 0: segment boundaries from sorted batch_ids (no atomics) ----
__global__ void __launch_bounds__(256) k_bounds(const int* __restrict__ bids, int N,
                                                int* __restrict__ firsts) {
  int n = blockIdx.x * 256 + threadIdx.x;
  if (n >= N) return;
  if (n == 0) { firsts[bids[0]] = 0; firsts[BGRAPHS] = N; }
  else if (bids[n] != bids[n - 1]) firsts[bids[n]] = n;
}

// ---- Kernel 1: segment max pool + q0/q1 = pooled @ Wq.T ----
__global__ void __launch_bounds__(256) k_pool_q(const float* __restrict__ inp,
                                                const int* __restrict__ firsts,
                                                const float* __restrict__ Wq0,
                                                const float* __restrict__ Wq1,
                                                float* __restrict__ qbuf) {
  __shared__ float pooled[DDIM];
  int b = blockIdx.x, t = threadIdx.x;
  int first = firsts[b], end = firsts[b + 1];
  float m = -3.0e38f;
  for (int n = first; n < end; ++n) m = fmaxf(m, inp[(size_t)n * DDIM + t]);
  pooled[t] = m;
  __syncthreads();
  for (int j = t; j < HEADS * KDIM; j += 256) {
    float s0 = 0.f, s1 = 0.f;
    for (int d = 0; d < DDIM; ++d) {
      float p = pooled[d];
      s0 += p * Wq0[j * DDIM + d];
      s1 += p * Wq1[j * DDIM + d];
    }
    qbuf[b * 1024 + j]       = s0;
    qbuf[b * 1024 + 512 + j] = s1;
  }
}

// ---- stage CHUNK nodes x 256 dims, row-major [node][dim] (Phase A) ----
__device__ __forceinline__ void stage_rm(const float* __restrict__ inp, int first, int count,
                                         int base, __bf16* xhi, __bf16* xlo, int tid) {
  for (int idx = tid; idx < CHUNK * DDIM; idx += 256) {
    int nl = idx >> 8;
    int n  = base + nl;
    float v = (n < count) ? inp[(size_t)(first + n) * DDIM + (idx & 255)] : 0.0f;
    __bf16 h = (__bf16)v;
    xhi[idx] = h;
    xlo[idx] = (__bf16)(v - (float)h);
  }
}

// ---- stage CHUNK nodes x 256 dims, transposed [dim][node], padded rows (Phase C) ----
__device__ __forceinline__ void stage_tr(const float* __restrict__ inp, int first, int count,
                                         int base, __bf16* xhi, __bf16* xlo, int tid) {
  for (int idx = tid; idx < CHUNK * DDIM; idx += 256) {
    int nl = idx >> 8, d = idx & 255;       // consecutive tids -> consecutive d: coalesced
    int n  = base + nl;
    float v = (n < count) ? inp[(size_t)(first + n) * DDIM + d] : 0.0f;
    __bf16 h = (__bf16)v;
    xhi[d * XTS + nl] = h;
    xlo[d * XTS + nl] = (__bf16)(v - (float)h);
  }
}

// ---- Kernel 2: per-graph attention (scores, softmax, att@x, embed, out) ----
__global__ void __launch_bounds__(256) k_attend(
    const float* __restrict__ inp, const int* __restrict__ firsts,
    const float* __restrict__ qbuf, const float* __restrict__ Wkeys,
    const float* __restrict__ Wembed, const float* __restrict__ Wout,
    const float* __restrict__ bout, float* __restrict__ out) {
  extern __shared__ __align__(16) char smem[];
  float*  sc    = (float*)(smem);                    // [16][2048] f32      131072 B
  __bf16* xhi   = (__bf16*)(smem + 131072);          // stage hi             69632 B
  __bf16* xlo   = (__bf16*)(smem + 200704);          // stage lo             69632 B
  __bf16* phi   = (__bf16*)(smem + 270336);          // p hi [16][256]        8192 B
  __bf16* plo   = (__bf16*)(smem + 278528);          // p lo                  8192 B
  float*  featb = (float*)(smem + 286720);           // feat [16][256]       16384 B
  float*  dxbuf = (float*)(smem + 303104);           // dx [256]              1024 B
  float*  red   = (float*)(smem + 304128);           // reduce [256]          1024 B
  unsigned* scu = (unsigned*)sc;                     // packed att (bf16 hi|lo)

  const int b = blockIdx.x, tid = threadIdx.x;
  const int wave = tid >> 5, lane = tid & 31;
  const int lm = lane & 15, half = lane >> 4;
  const int first = firsts[b];
  const int count = firsts[b + 1] - first;

  for (int br = 0; br < 2; ++br) {
    // p[h][d] = sum_k q[h*32+k] * Wkeys[k][d], split hi/lo bf16
    const float* q = qbuf + b * 1024 + br * 512;
    for (int idx = tid; idx < HEADS * DDIM; idx += 256) {
      int h = idx >> 8, d = idx & 255;
      float s = 0.f;
      #pragma unroll
      for (int k = 0; k < KDIM; ++k) s += q[h * KDIM + k] * Wkeys[k * DDIM + d];
      __bf16 hb = (__bf16)s;
      phi[idx] = hb;
      plo[idx] = (__bf16)(s - (float)hb);
    }
    __syncthreads();

    // ---- Phase A: scores[h][n] = p[h] . x[n] via WMMA (3-term bf16 split) ----
    for (int base = 0; base < count; base += CHUNK) {
      stage_rm(inp, first, count, base, xhi, xlo, tid);
      __syncthreads();
      v8f c = {0.f, 0.f, 0.f, 0.f, 0.f, 0.f, 0.f, 0.f};
      const int nrow = wave * 16 + lm;   // node (B column) for this lane
      #pragma unroll
      for (int dc = 0; dc < 8; ++dc) {
        const int dbase = dc * 32;
        v16bf pa, pl, bh, bl;
        #pragma unroll
        for (int v = 0; v < 8; ++v) {
          int ka = ((v >> 2) << 4) + (half << 3) + ((v & 3) << 1);   // A layout
          pa[2 * v]     = phi[lm * DDIM + dbase + ka];
          pa[2 * v + 1] = phi[lm * DDIM + dbase + ka + 1];
          pl[2 * v]     = plo[lm * DDIM + dbase + ka];
          pl[2 * v + 1] = plo[lm * DDIM + dbase + ka + 1];
          int kb = ((v >> 2) << 4) + (half << 3) + ((v & 3) << 1);   // B layout
          bh[2 * v]     = xhi[nrow * DDIM + dbase + kb];
          bh[2 * v + 1] = xhi[nrow * DDIM + dbase + kb + 1];
          bl[2 * v]     = xlo[nrow * DDIM + dbase + kb];
          bl[2 * v + 1] = xlo[nrow * DDIM + dbase + kb + 1];
        }
        c = wmma_bf(pa, bh, c);
        c = wmma_bf(pa, bl, c);
        c = wmma_bf(pl, bh, c);
      }
      #pragma unroll
      for (int r = 0; r < 8; ++r)                                    // C layout
        sc[(r + (half << 3)) * MAXN + base + wave * 16 + lm] = c[r];
      __syncthreads();
    }

    // ---- Phase B: softmax per head; normalize and pack att as bf16 hi|lo ----
    {
      const int h = tid >> 4, i = tid & 15;
      float mx = -3.0e38f;
      for (int n = i; n < count; n += 16) mx = fmaxf(mx, sc[h * MAXN + n]);
      red[tid] = mx;
      __syncthreads();
      float gmx = -3.0e38f;
      #pragma unroll
      for (int j = 0; j < 16; ++j) gmx = fmaxf(gmx, red[h * 16 + j]);
      float sum = 0.f;
      for (int n = i; n < count; n += 16) {
        float e = __expf(sc[h * MAXN + n] - gmx);
        sc[h * MAXN + n] = e;
        sum += e;
      }
      __syncthreads();
      red[tid] = sum;
      __syncthreads();
      float gs = 0.f;
      #pragma unroll
      for (int j = 0; j < 16; ++j) gs += red[h * 16 + j];
      float inv = 1.0f / gs;
      for (int n = i; n < count; n += 16) {
        float a = sc[h * MAXN + n] * inv;
        __bf16 hh = (__bf16)a;
        __bf16 ll = (__bf16)(a - (float)hh);
        scu[h * MAXN + n] = (unsigned)__builtin_bit_cast(unsigned short, hh) |
                            ((unsigned)__builtin_bit_cast(unsigned short, ll) << 16);
      }
    }

    // ---- Phase C: feat[h][d] = sum_n att[h][n]*x[n][d] via WMMA ----
    v8f a0 = {0.f, 0.f, 0.f, 0.f, 0.f, 0.f, 0.f, 0.f};
    v8f a1 = {0.f, 0.f, 0.f, 0.f, 0.f, 0.f, 0.f, 0.f};
    const int dt0 = wave, dt1 = wave + 8;    // this wave's two 16-wide dim tiles
    for (int base = 0; base < count; base += CHUNK) {
      stage_tr(inp, first, count, base, xhi, xlo, tid);
      __syncthreads();
      int rem = count - base;
      int kcmax = ((rem < CHUNK ? rem : CHUNK) + 31) >> 5;   // uniform
      for (int kc = 0; kc < kcmax; ++kc) {
        // A fragment: packed att, unpack hi|lo (contiguous dwords -> b128 loads)
        v16bf ah, al;
        #pragma unroll
        for (int v = 0; v < 8; ++v) {
          #pragma unroll
          for (int j = 0; j < 2; ++j) {
            int ka = (half << 4) + (v << 1) + j;  // A: k = node
            int n = base + kc * 32 + ka;
            unsigned w = (n < count) ? scu[lm * MAXN + n] : 0u;
            ah[2 * v + j] = __builtin_bit_cast(__bf16, (unsigned short)(w & 0xFFFFu));
            al[2 * v + j] = __builtin_bit_cast(__bf16, (unsigned short)(w >> 16));
          }
        }
        // tile 0: build B, consume, then reuse the same regs for tile 1
        {
          v16bf bh, bl;
          const int rowb = (dt0 * 16 + lm) * XTS + kc * 32 + (half << 4);
          #pragma unroll
          for (int v = 0; v < 8; ++v) {
            bh[2 * v]     = xhi[rowb + 2 * v];
            bh[2 * v + 1] = xhi[rowb + 2 * v + 1];
            bl[2 * v]     = xlo[rowb + 2 * v];
            bl[2 * v + 1] = xlo[rowb + 2 * v + 1];
          }
          a0 = wmma_bf(ah, bh, a0);
          a0 = wmma_bf(ah, bl, a0);
          a0 = wmma_bf(al, bh, a0);
        }
        {
          v16bf bh, bl;
          const int rowb = (dt1 * 16 + lm) * XTS + kc * 32 + (half << 4);
          #pragma unroll
          for (int v = 0; v < 8; ++v) {
            bh[2 * v]     = xhi[rowb + 2 * v];
            bh[2 * v + 1] = xhi[rowb + 2 * v + 1];
            bl[2 * v]     = xlo[rowb + 2 * v];
            bl[2 * v + 1] = xlo[rowb + 2 * v + 1];
          }
          a1 = wmma_bf(ah, bh, a1);
          a1 = wmma_bf(ah, bl, a1);
          a1 = wmma_bf(al, bh, a1);
        }
      }
      __syncthreads();
    }
    #pragma unroll
    for (int r = 0; r < 8; ++r) {
      featb[(r + (half << 3)) * DDIM + dt0 * 16 + lm] = a0[r];
      featb[(r + (half << 3)) * DDIM + dt1 * 16 + lm] = a1[r];
    }
    __syncthreads();

    // ---- e[h][r] = feat[h] . Wembed[r];  dx = e0 - e1 ----
    {
      const int h = tid >> 4, r = tid & 15;
      float s = 0.f;
      for (int d = 0; d < DDIM; ++d) s += featb[h * DDIM + d] * Wembed[r * DDIM + d];
      if (br == 0) dxbuf[tid] = s; else dxbuf[tid] -= s;
    }
    __syncthreads();
  }

  // ---- out[b][j] = leaky_relu(Wout[j] . dx + bout[j]) ----
  float s = bout[tid];
  for (int i = 0; i < DDIM; ++i) s += Wout[tid * DDIM + i] * dxbuf[i];
  out[b * DDIM + tid] = s > 0.f ? s : 0.01f * s;
}

extern "C" void kernel_launch(void* const* d_in, const int* in_sizes, int n_in,
                              void* d_out, int out_size, void* d_ws, size_t ws_size,
                              hipStream_t stream) {
  const float* inp   = (const float*)d_in[0];
  const int*   bids  = (const int*)d_in[1];
  const float* Wkeys = (const float*)d_in[4];
  const float* Wq0   = (const float*)d_in[5];
  const float* Wq1   = (const float*)d_in[6];
  const float* Wemb  = (const float*)d_in[7];
  const float* Wout  = (const float*)d_in[8];
  const float* bout  = (const float*)d_in[9];
  float* out = (float*)d_out;

  const int N = in_sizes[1];
  int*   firsts = (int*)d_ws;                       // (B+1) ints
  float* qbuf   = (float*)((char*)d_ws + 512);      // [B][2][512] floats

  k_bounds<<<(N + 255) / 256, 256, 0, stream>>>(bids, N, firsts);
  k_pool_q<<<BGRAPHS, 256, 0, stream>>>(inp, firsts, Wq0, Wq1, qbuf);
  k_attend<<<BGRAPHS, 256, 305152, stream>>>(inp, firsts, qbuf, Wkeys, Wemb, Wout, bout, out);
}
